// TopologicalConvTransformer_26972394619323
// MI455X (gfx1250) — matrix operand order repaired and checked by reading, not compiled, not run on hardware
//
#include <hip/hip_runtime.h>
#include <hip/hip_bf16.h>

typedef unsigned int u32;
typedef __attribute__((ext_vector_type(16))) _Float16 v16h;
typedef __attribute__((ext_vector_type(8)))  float    v8f;

#define B_    16
#define CIN   3
#define HW_   224
#define OC    16
#define OHW   222
#define NPB   (OC*OHW*OHW)       /* 788544 conv outputs per batch */
#define SEQ   1024
#define HID   64
#define NHEAD_ 2
#define HD    32
#define DFF_  16
#define NL    2
#define LN_EPS_ 1e-5f

/* ---- monotone float<->uint key mapping (ascending order preserved) ---- */
static __device__ __forceinline__ u32 enc_key(float f) {
  u32 u = __float_as_uint(f);
  return (u & 0x80000000u) ? ~u : (u | 0x80000000u);
}
static __device__ __forceinline__ float dec_key(u32 u) {
  return (u & 0x80000000u) ? __uint_as_float(u ^ 0x80000000u)
                           : __uint_as_float(~u);
}

/* ---- WMMA fragment helpers (wave32, 16x16x32 f16 -> f32) ----
   A 16x32: lane holds row m=lane&15; elem e -> K = (lane>>4)*16 + e.
   B 32x16 (loaded as X @ W^T, i.e. B[k][n] = W[n][k], contiguous in k):
            lane holds col n=lane&15; elem e -> K = (lane>>4)*16 + e.
   C 16x16 f32 (v8f): row = r + (lane>>4)*8, col = lane&15.              */
static __device__ __forceinline__ v16h load_a16(const _Float16* base, int ld) {
  int lane = threadIdx.x & 31;
  const _Float16* p = base + (lane & 15) * ld + ((lane >> 4) * 16);
  v16h a;
#pragma unroll
  for (int e = 0; e < 16; ++e) a[e] = p[e];
  return a;
}
static __device__ __forceinline__ v16h load_bT16(const _Float16* w, int ld) {
  int lane = threadIdx.x & 31;
  const _Float16* p = w + (lane & 15) * ld + ((lane >> 4) * 16);
  v16h b;
#pragma unroll
  for (int e = 0; e < 16; ++e) b[e] = p[e];
  return b;
}
#define WMMA_F32_F16(a,b,c) \
  __builtin_amdgcn_wmma_f32_16x16x32_f16(false,(a),false,(b),(short)0,(c),false,false)

/* ================= conv 3x3 VALID + bias -> ordered keys ================= */
__global__ void k_conv(const float* __restrict__ xs, const float* __restrict__ cw,
                       const float* __restrict__ cb, u32* __restrict__ keys) {
  long long i = (long long)blockIdx.x * blockDim.x + threadIdx.x;
  if (i >= (long long)B_ * NPB) return;
  int b  = (int)(i / NPB);
  int r  = (int)(i % NPB);
  int oc = r / (OHW * OHW);
  int rr = r % (OHW * OHW);
  int oy = rr / OHW, ox = rr % OHW;
  const float inv256 = 1.0f / 256.0f;
  float acc = cb[oc];
#pragma unroll
  for (int ic = 0; ic < CIN; ++ic)
#pragma unroll
    for (int ky = 0; ky < 3; ++ky)
#pragma unroll
      for (int kx = 0; kx < 3; ++kx)
        acc += xs[((b * CIN + ic) * HW_ + oy + ky) * HW_ + ox + kx] * inv256 *
               cw[((oc * CIN + ic) * 3 + ky) * 3 + kx];
  keys[(long long)b * NPB + r] = enc_key(acc);
}

/* ==== per-batch radix-select of 2048 smallest + LDS bitonic sort ==== */
__global__ void __launch_bounds__(1024) k_select(const u32* __restrict__ keys,
                                                 float* __restrict__ bd) {
  __shared__ u32 hist[256];
  __shared__ u32 sel[2048];
  __shared__ u32 s_sel, s_below, s_cl, s_ce;
  int b = blockIdx.x, tid = threadIdx.x;
  const u32* kb = keys + (long long)b * NPB;

  u32 prefix = 0, rank = 2047, less_total = 0;
  for (int round = 0; round < 4; ++round) {
    int shift = 24 - 8 * round;
    u32 mask = (round == 0) ? 0u : (0xFFFFFFFFu << (32 - 8 * round));
    for (int j = tid; j < 256; j += 1024) hist[j] = 0;
    __syncthreads();
    for (int j = tid; j < NPB; j += 1024) {
      u32 v = kb[j];
      if ((v & mask) == prefix) atomicAdd(&hist[(v >> shift) & 255], 1u);
    }
    __syncthreads();
    if (tid == 0) {
      u32 acc = 0, s = 255, below = 0;
      for (u32 j = 0; j < 256; ++j) {
        if (acc + hist[j] > rank) { s = j; below = acc; break; }
        acc += hist[j];
      }
      s_sel = s; s_below = below;
    }
    __syncthreads();
    prefix |= s_sel << shift;
    rank -= s_below;
    less_total += s_below;
    __syncthreads();
  }
  u32 T = prefix;            /* 2048-th smallest key; less_total keys < T */
  if (tid == 0) { s_cl = 0; s_ce = 0; }
  __syncthreads();
  for (int j = tid; j < NPB; j += 1024) {
    u32 v = kb[j];
    if (v < T)       { u32 p = atomicAdd(&s_cl, 1u); if (p < 2048) sel[p] = v; }
    else if (v == T) { u32 p = atomicAdd(&s_ce, 1u); u32 q = less_total + p;
                       if (q < 2048) sel[q] = v; }
  }
  __syncthreads();
  /* bitonic sort of 2048 keys in LDS */
  for (int k = 2; k <= 2048; k <<= 1)
    for (int j = k >> 1; j > 0; j >>= 1) {
      for (int idx = tid; idx < 2048; idx += 1024) {
        int ixj = idx ^ j;
        if (ixj > idx) {
          u32 a = sel[idx], c = sel[ixj];
          bool up = ((idx & k) == 0);
          if ((a > c) == up) { sel[idx] = c; sel[ixj] = a; }
        }
      }
      __syncthreads();
    }
  /* births = even positions, deaths = odd positions */
  bd[(long long)b * 2048 + tid]        = dec_key(sel[2 * tid]);
  bd[(long long)b * 2048 + 1024 + tid] = dec_key(sel[2 * tid + 1]);
}

/* ================= embedding: seq(b,d,0,0) @ emb_w^T + emb_b ============ */
__global__ void k_embed(const float* __restrict__ bd, const float* __restrict__ ew,
                        const float* __restrict__ eb, float* __restrict__ h,
                        _Float16* __restrict__ h16) {
  int i = blockIdx.x * blockDim.x + threadIdx.x;
  if (i >= B_ * SEQ * HID) return;
  int j = i & 63;
  int t = (i >> 6) & 1023;
  int b = i >> 16;
  float birth = bd[b * 2048 + t];
  float death = bd[b * 2048 + 1024 + t];
  float v = birth * ew[j * 4 + 0] + death * ew[j * 4 + 1] + eb[j];
  h[i] = v;
  h16[i] = (_Float16)v;
}

__global__ void k_f2h(const float* __restrict__ s, _Float16* __restrict__ d, int n) {
  int i = blockIdx.x * blockDim.x + threadIdx.x;
  if (i < n) d[i] = (_Float16)s[i];
}

/* ===== QKV projection: h @ wqkv^T + b (WMMA); V also stored transposed ==
   vT layout: vT[((b*2 + head)*32 + d) * SEQ + t]  (t contiguous)          */
__global__ void __launch_bounds__(128) k_qkv(const _Float16* __restrict__ h16,
                                             const _Float16* __restrict__ wq,
                                             const float* __restrict__ bq,
                                             _Float16* __restrict__ qkv16,
                                             _Float16* __restrict__ vT) {
  int wid = blockIdx.x * 4 + (threadIdx.x >> 5);
  int ot = wid % 12;
  int st = (wid / 12) & 63;
  int b  = wid / (12 * 64);
  int lane = threadIdx.x & 31;
  const _Float16* abase = h16 + ((long long)b * SEQ + st * 16) * HID;
  const _Float16* wbase = wq + ot * 16 * HID;
  v8f c = {};
#pragma unroll
  for (int kc = 0; kc < HID; kc += 32) {
    v16h a = load_a16(abase + kc, HID);
    v16h w = load_bT16(wbase + kc, HID);
    c = WMMA_F32_F16(a, w, c);
  }
  int col = lane & 15, rb = (lane >> 4) * 8;
  float bias = bq[ot * 16 + col];
  _Float16* out = qkv16 + ((long long)b * SEQ + st * 16) * 192 + ot * 16 + col;
#pragma unroll
  for (int r = 0; r < 8; ++r)
    out[(rb + r) * 192] = (_Float16)(c[r] + bias);
  if (ot >= 8) {                          /* V tile: also store transposed */
    int feat = ot * 16 + col - 128;       /* 0..63 */
    int head = feat >> 5, d = feat & 31;
    _Float16* vt = vT + (((long long)b * NHEAD_ + head) * HD + d) * SEQ
                      + st * 16 + rb;
#pragma unroll
    for (int r = 0; r < 8; ++r) vt[r] = (_Float16)(c[r] + bias);
  }
}

/* ====== flash attention: 1 wave per (batch, head, 16-row query tile) ==== */
__global__ void __launch_bounds__(32) k_attn(const _Float16* __restrict__ qkv16,
                                             const _Float16* __restrict__ vT,
                                             _Float16* __restrict__ ao16) {
  __shared__ _Float16 pls[16 * 32];   /* P tile staged C-layout -> A-layout */
  int qt   = blockIdx.x & 63;
  int head = (blockIdx.x >> 6) & 1;
  int b    = blockIdx.x >> 7;
  int lane = threadIdx.x & 31;
  int col = lane & 15, rb = (lane >> 4) * 8;
  const _Float16* base = qkv16 + (long long)b * SEQ * 192;
  const _Float16* vTb  = vT + ((long long)b * NHEAD_ + head) * HD * SEQ;
  v16h aq = load_a16(base + (qt * 16) * 192 + head * HD, 192);
  v8f o0 = {}, o1 = {};
  float m[8], lsum[8];
#pragma unroll
  for (int r = 0; r < 8; ++r) { m[r] = -__builtin_inff(); lsum[r] = 0.f; }
  const float scale = 0.1767766952966369f;  /* 1/sqrt(32) */

  for (int jt = 0; jt < 64; jt += 2) {
    /* scores vs 32 keys: K dim = head_dim 32 (one WMMA per 16-key tile) */
    v16h bk0 = load_bT16(base + (jt * 16) * 192 + 64 + head * HD, 192);
    v16h bk1 = load_bT16(base + ((jt + 1) * 16) * 192 + 64 + head * HD, 192);
    v8f z = {};
    v8f s0 = WMMA_F32_F16(aq, bk0, z);
    v8f s1 = WMMA_F32_F16(aq, bk1, z);
#pragma unroll
    for (int r = 0; r < 8; ++r) {
      s0[r] *= scale; s1[r] *= scale;
      float tmax = fmaxf(s0[r], s1[r]);
#pragma unroll
      for (int d = 1; d < 16; d <<= 1) tmax = fmaxf(tmax, __shfl_xor(tmax, d, 32));
      float nm = fmaxf(m[r], tmax);
      float corr = __expf(m[r] - nm);
      float p0 = __expf(s0[r] - nm);
      float p1 = __expf(s1[r] - nm);
      float rs = p0 + p1;
#pragma unroll
      for (int d = 1; d < 16; d <<= 1) rs += __shfl_xor(rs, d, 32);
      lsum[r] = lsum[r] * corr + rs;
      o0[r] *= corr; o1[r] *= corr;
      m[r] = nm;
      pls[(rb + r) * 32 + col]      = (_Float16)p0;
      pls[(rb + r) * 32 + 16 + col] = (_Float16)p1;
    }
    __syncthreads();
    v16h ap = load_a16(pls, 32);            /* P as A-fragment (16x32)     */
    /* V as B: B[k=t'][n=d] = vT[d][jt*16 + k] -> contiguous in k (b128s)  */
    v16h bv0 = load_bT16(vTb + jt * 16, SEQ);
    v16h bv1 = load_bT16(vTb + 16 * SEQ + jt * 16, SEQ);
    o0 = WMMA_F32_F16(ap, bv0, o0);
    o1 = WMMA_F32_F16(ap, bv1, o1);
    __syncthreads();
  }
  _Float16* out = ao16 + ((long long)b * SEQ + qt * 16) * HID + head * HD;
#pragma unroll
  for (int r = 0; r < 8; ++r) {
    float inv = 1.0f / lsum[r];
    out[(rb + r) * HID + col]      = (_Float16)(o0[r] * inv);
    out[(rb + r) * HID + 16 + col] = (_Float16)(o1[r] * inv);
  }
}

/* ========== output projection + residual + LayerNorm1 (WMMA) =========== */
__global__ void __launch_bounds__(128) k_proj_ln(const _Float16* __restrict__ ao16,
                     const _Float16* __restrict__ wo, const float* __restrict__ bo,
                     const float* __restrict__ ln_s, const float* __restrict__ ln_b,
                     float* __restrict__ h, _Float16* __restrict__ h16) {
  __shared__ float sh[16 * 64];
  int st = blockIdx.x & 63;
  int b  = blockIdx.x >> 6;
  int wv = threadIdx.x >> 5;       /* 0..3: 16-col output tile */
  int lane = threadIdx.x & 31;
  const _Float16* abase = ao16 + ((long long)b * SEQ + st * 16) * HID;
  v8f c = {};
#pragma unroll
  for (int kc = 0; kc < HID; kc += 32) {
    v16h a = load_a16(abase + kc, HID);
    v16h w = load_bT16(wo + wv * 16 * HID + kc, HID);
    c = WMMA_F32_F16(a, w, c);
  }
  int col = lane & 15, rb = (lane >> 4) * 8;
  float bias = bo[wv * 16 + col];
#pragma unroll
  for (int r = 0; r < 8; ++r) sh[(rb + r) * 64 + wv * 16 + col] = c[r] + bias;
  __syncthreads();
  if (threadIdx.x < 16) {
    int row = threadIdx.x;
    float* hrow = h + ((long long)b * SEQ + st * 16 + row) * HID;
    _Float16* h16row = h16 + ((long long)b * SEQ + st * 16 + row) * HID;
    float mean = 0.f;
    for (int cix = 0; cix < 64; ++cix) mean += hrow[cix] + sh[row * 64 + cix];
    mean *= (1.f / 64.f);
    float var = 0.f;
    for (int cix = 0; cix < 64; ++cix) {
      float d = hrow[cix] + sh[row * 64 + cix] - mean; var += d * d;
    }
    float inv = rsqrtf(var * (1.f / 64.f) + LN_EPS_);
    for (int cix = 0; cix < 64; ++cix) {
      float o = (hrow[cix] + sh[row * 64 + cix] - mean) * inv * ln_s[cix] + ln_b[cix];
      hrow[cix] = o;
      h16row[cix] = (_Float16)o;
    }
  }
}

/* ========== FFN (relu(h@W1^T+b1)@W2^T+b2) + residual + LN2 ============= */
__global__ void __launch_bounds__(32) k_ff_ln(const _Float16* __restrict__ w1,
        const float* __restrict__ b1, const _Float16* __restrict__ w2,
        const float* __restrict__ b2, const float* __restrict__ ln_s,
        const float* __restrict__ ln_b, float* __restrict__ h,
        _Float16* __restrict__ h16) {
  __shared__ _Float16 shp[16 * 32];
  __shared__ float shf[16 * 64];
  int st = blockIdx.x & 63;
  int b  = blockIdx.x >> 6;
  int lane = threadIdx.x & 31;
  int col = lane & 15, rb = (lane >> 4) * 8;
  const _Float16* abase = h16 + ((long long)b * SEQ + st * 16) * HID;
  v8f c = {};
#pragma unroll
  for (int kc = 0; kc < HID; kc += 32) {
    v16h a = load_a16(abase + kc, HID);
    v16h w = load_bT16(w1 + kc, HID);         /* W1: (16,64) rows */
    c = WMMA_F32_F16(a, w, c);
  }
#pragma unroll
  for (int r = 0; r < 8; ++r) {
    float v = fmaxf(c[r] + b1[col], 0.f);
    shp[(rb + r) * 32 + col] = (_Float16)v;   /* cols 0..15 = hidden  */
    shp[(rb + r) * 32 + 16 + col] = (_Float16)0.f;  /* zero-pad K 16..31 */
  }
  __syncthreads();
  v16h ap = load_a16(shp, 32);
#pragma unroll
  for (int ot = 0; ot < 4; ++ot) {
    v16h bw;
#pragma unroll
    for (int e = 0; e < 16; ++e)
      bw[e] = (lane < 16) ? w2[(ot * 16 + col) * DFF_ + e] : (_Float16)0.f;
    v8f z = {};
    v8f c2 = WMMA_F32_F16(ap, bw, z);
#pragma unroll
    for (int r = 0; r < 8; ++r)
      shf[(rb + r) * 64 + ot * 16 + col] = c2[r] + b2[ot * 16 + col];
  }
  __syncthreads();
  if (lane < 16) {
    int row = lane;
    float* hrow = h + ((long long)b * SEQ + st * 16 + row) * HID;
    _Float16* h16row = h16 + ((long long)b * SEQ + st * 16 + row) * HID;
    float mean = 0.f;
    for (int cix = 0; cix < 64; ++cix) mean += hrow[cix] + shf[row * 64 + cix];
    mean *= (1.f / 64.f);
    float var = 0.f;
    for (int cix = 0; cix < 64; ++cix) {
      float d = hrow[cix] + shf[row * 64 + cix] - mean; var += d * d;
    }
    float inv = rsqrtf(var * (1.f / 64.f) + LN_EPS_);
    for (int cix = 0; cix < 64; ++cix) {
      float o = (hrow[cix] + shf[row * 64 + cix] - mean) * inv * ln_s[cix] + ln_b[cix];
      hrow[cix] = o;
      h16row[cix] = (_Float16)o;
    }
  }
}

/* ========== mean pool over hidden, classifier, softmax ================== */
__global__ void __launch_bounds__(256) k_head(const float* __restrict__ h,
                       const float* __restrict__ cw, const float* __restrict__ cb,
                       float* __restrict__ out) {
  __shared__ float pooled[SEQ];
  __shared__ float logits[16];
  int b = blockIdx.x, tid = threadIdx.x;
  for (int t = tid; t < SEQ; t += 256) {
    const float* row = h + ((long long)b * SEQ + t) * HID;
    float s = 0.f;
    for (int c2 = 0; c2 < HID; ++c2) s += row[c2];
    pooled[t] = s * (1.0f / 64.0f);
  }
  __syncthreads();
  if (tid < 10) {
    float s = cb[tid];
    for (int t = 0; t < SEQ; ++t) s += pooled[t] * cw[tid * SEQ + t];
    logits[tid] = s;
  }
  __syncthreads();
  if (tid == 0) {
    float mx = logits[0];
    for (int o = 1; o < 10; ++o) mx = fmaxf(mx, logits[o]);
    float e[10], sum = 0.f;
    for (int o = 0; o < 10; ++o) { e[o] = __expf(logits[o] - mx); sum += e[o]; }
    for (int o = 0; o < 10; ++o) out[b * 10 + o] = e[o] / sum;
  }
}

extern "C" void kernel_launch(void* const* d_in, const int* in_sizes, int n_in,
                              void* d_out, int out_size, void* d_ws, size_t ws_size,
                              hipStream_t stream) {
  const float* xs     = (const float*)d_in[0];
  const float* conv_w = (const float*)d_in[1];
  const float* conv_b = (const float*)d_in[2];
  const float* emb_w  = (const float*)d_in[3];
  const float* emb_b  = (const float*)d_in[4];
  const float* qkv_w  = (const float*)d_in[5];
  const float* qkv_b  = (const float*)d_in[6];
  const float* out_w  = (const float*)d_in[7];
  const float* out_b  = (const float*)d_in[8];
  const float* ln1_s  = (const float*)d_in[9];
  const float* ln1_b  = (const float*)d_in[10];
  const float* ff1_w  = (const float*)d_in[11];
  const float* ff1_b  = (const float*)d_in[12];
  const float* ff2_w  = (const float*)d_in[13];
  const float* ff2_b  = (const float*)d_in[14];
  const float* ln2_s  = (const float*)d_in[15];
  const float* ln2_b  = (const float*)d_in[16];
  const float* cls_w  = (const float*)d_in[17];
  const float* cls_b  = (const float*)d_in[18];
  float* out = (float*)d_out;
  char* ws = (char*)d_ws;
  (void)in_sizes; (void)n_in; (void)out_size; (void)ws_size;

  size_t off = 0;
  auto alloc = [&](size_t bytes) {
    void* p = ws + off;
    off = (off + bytes + 255) & ~(size_t)255;
    return p;
  };
  u32*      keys  = (u32*)alloc(sizeof(u32) * (size_t)B_ * NPB);
  float*    bd    = (float*)alloc(sizeof(float) * B_ * 2048);
  float*    h     = (float*)alloc(sizeof(float) * (size_t)B_ * SEQ * HID);
  _Float16* h16   = (_Float16*)alloc(sizeof(_Float16) * (size_t)B_ * SEQ * HID);
  _Float16* qkv16 = (_Float16*)alloc(sizeof(_Float16) * (size_t)B_ * SEQ * 192);
  _Float16* vT    = (_Float16*)alloc(sizeof(_Float16) * (size_t)B_ * NHEAD_ * HD * SEQ);
  _Float16* ao16  = (_Float16*)alloc(sizeof(_Float16) * (size_t)B_ * SEQ * HID);
  _Float16* qkvW  = (_Float16*)alloc(sizeof(_Float16) * NL * 192 * HID);
  _Float16* outW  = (_Float16*)alloc(sizeof(_Float16) * NL * HID * HID);
  _Float16* ff1W  = (_Float16*)alloc(sizeof(_Float16) * NL * DFF_ * HID);
  _Float16* ff2W  = (_Float16*)alloc(sizeof(_Float16) * NL * HID * DFF_);

  /* weight conversion to f16 */
  k_f2h<<<(NL * 192 * HID + 255) / 256, 256, 0, stream>>>(qkv_w, qkvW, NL * 192 * HID);
  k_f2h<<<(NL * HID * HID + 255) / 256, 256, 0, stream>>>(out_w, outW, NL * HID * HID);
  k_f2h<<<(NL * DFF_ * HID + 255) / 256, 256, 0, stream>>>(ff1_w, ff1W, NL * DFF_ * HID);
  k_f2h<<<(NL * HID * DFF_ + 255) / 256, 256, 0, stream>>>(ff2_w, ff2W, NL * HID * DFF_);

  long long nconv = (long long)B_ * NPB;
  k_conv<<<(int)((nconv + 255) / 256), 256, 0, stream>>>(xs, conv_w, conv_b, keys);
  k_select<<<B_, 1024, 0, stream>>>(keys, bd);
  k_embed<<<(B_ * SEQ * HID + 255) / 256, 256, 0, stream>>>(bd, emb_w, emb_b, h, h16);

  for (int l = 0; l < NL; ++l) {
    k_qkv<<<(B_ * 64 * 12) / 4, 128, 0, stream>>>(h16, qkvW + l * 192 * HID,
                                                  qkv_b + l * 192, qkv16, vT);
    k_attn<<<B_ * NHEAD_ * 64, 32, 0, stream>>>(qkv16, vT, ao16);
    k_proj_ln<<<B_ * 64, 128, 0, stream>>>(ao16, outW + l * HID * HID, out_b + l * HID,
                                           ln1_s + l * HID, ln1_b + l * HID, h, h16);
    k_ff_ln<<<B_ * 64, 32, 0, stream>>>(ff1W + l * DFF_ * HID, ff1_b + l * DFF_,
                                        ff2W + l * HID * DFF_, ff2_b + l * HID,
                                        ln2_s + l * HID, ln2_b + l * HID, h, h16);
  }
  k_head<<<B_, 256, 0, stream>>>(h, cls_w, cls_b, out);
}
